// CRF_62362925138769
// MI455X (gfx1250) — matrix operand order repaired and checked by reading, not compiled, not run on hardware
//
#include <hip/hip_runtime.h>
#include <hip/hip_fp16.h>

#define N_STATES 1024
#define N_OBS    32000
#define SEQ_LEN  4096

#define FWD_WGS      64     // one 16-column tile per workgroup
#define FWD_THREADS  128    // 4 waves of 32
#define COLS_PER_WG  16

typedef __attribute__((ext_vector_type(16))) _Float16 v16h;
typedef __attribute__((ext_vector_type(8)))  _Float16 v8h;
typedef __attribute__((ext_vector_type(8)))  float    v8f;

// ---------------------------------------------------------------------------
// Kernel: zero the grid-barrier counter (must run before forward kernel each
// launch so graph replays are deterministic).
// ---------------------------------------------------------------------------
__global__ void crf_init_kernel(unsigned long long* bar) {
    if (threadIdx.x == 0) *bar = 0ull;
}

// ---------------------------------------------------------------------------
// Kernel: E = exp(transition) in fp16 (row-major [k][j]).
// T ~ N(-1, 0.1) so exp(T) in ~[0.2, 0.6]: safely fp16.
// ---------------------------------------------------------------------------
__global__ void crf_exp_kernel(const float* __restrict__ T, _Float16* __restrict__ E) {
    int i = blockIdx.x * 256 + threadIdx.x;
    if (i < N_STATES * N_STATES) E[i] = (_Float16)__expf(T[i]);
}

// ---------------------------------------------------------------------------
// Kernel: gather em[t][j] = emission[j, obs[t]]  (dense [SEQ_LEN][N_STATES]).
// Distinct (j, obs) lines ~ 254 MB of HBM => ~11 us at 23.3 TB/s.
// ---------------------------------------------------------------------------
__global__ void crf_gather_kernel(const float* __restrict__ emis,
                                  const int* __restrict__ obs,
                                  float* __restrict__ em) {
    int t = blockIdx.x;
    int o = obs[t];
    for (int j = threadIdx.x; j < N_STATES; j += 256)
        em[(size_t)t * N_STATES + j] = emis[(size_t)j * N_OBS + o];
}

// ---------------------------------------------------------------------------
// Kernel: log-numerator (path score) -> ws_num[0].
// ---------------------------------------------------------------------------
__global__ void crf_numer_kernel(const float* __restrict__ start,
                                 const float* __restrict__ trans,
                                 const float* __restrict__ emis,
                                 const int* __restrict__ obs,
                                 const int* __restrict__ st,
                                 float* __restrict__ outv) {
    __shared__ float red[256];
    float acc = 0.f;
    for (int t = threadIdx.x; t < SEQ_LEN; t += 256) {
        acc += emis[(size_t)st[t] * N_OBS + obs[t]];
        if (t >= 1) acc += trans[(size_t)st[t - 1] * N_STATES + st[t]];
    }
    red[threadIdx.x] = acc;
    __syncthreads();
    for (int s = 128; s > 0; s >>= 1) {
        if (threadIdx.x < s) red[threadIdx.x] += red[threadIdx.x + s];
        __syncthreads();
    }
    if (threadIdx.x == 0) outv[0] = red[0] + start[st[0]];
}

// ---------------------------------------------------------------------------
// Kernel: forward recursion (persistent, grid-synchronized), WMMA matvec.
//   alpha_t (fp32) double-buffered in global; v = exp(alpha-m) in fp16 LDS;
//   E column-slab (1024 x 16 fp16) pre-swizzled into B-fragment layout in LDS.
// ---------------------------------------------------------------------------
__global__ void __launch_bounds__(FWD_THREADS)
crf_forward_kernel(const float* __restrict__ start,
                   const _Float16* __restrict__ E,
                   const float* __restrict__ em,
                   const float* __restrict__ ws_num,
                   unsigned long long* __restrict__ bar,
                   float* __restrict__ alphaG,     // [2][N_STATES]
                   float* __restrict__ out) {
    // B fragments: [kk=0..31][lane=0..31][16 halves] = 32 KB
    __shared__ __align__(32) _Float16 sB[32 * 32 * 16];
    __shared__ __align__(16) _Float16 sV[N_STATES];   // exp(alpha - m), fp16
    __shared__ float sA[N_STATES];                    // current alpha, fp32
    __shared__ float sRed[4];                         // per-wave partial reduce
    __shared__ float sPart[4][16];

    const int tid   = threadIdx.x;
    const int wg    = blockIdx.x;
    const int lane  = tid & 31;
    const int wave  = tid >> 5;
    const int cbase = wg * COLS_PER_WG;

    // ---- Load & swizzle this WG's 1024x16 E-slab into B-fragment layout ----
    // B packing used here (contiguous k, matches A): lane L<16  -> col=L,    K = kk*32 + (0..15)
    //                         lane L>=16 -> col=L-16, K = kk*32 + (16..31)
    for (int i = tid; i < 32 * 32 * 16; i += FWD_THREADS) {
        int kk = i >> 9;
        int L  = (i >> 4) & 31;
        int h  = i & 15;
        int k  = kk * 32 + ((L >> 4) << 4) + h;
        int c  = cbase + (L & 15);
        sB[i] = E[(size_t)k * N_STATES + c];
    }

    // ---- Seed alpha_0 = start + em[0] (buffer 0) ----
    if (tid < COLS_PER_WG) {
        int j = cbase + tid;
        alphaG[j] = start[j] + em[j];
    }

    unsigned long long nb = 1;

    // grid-wide barrier: monotonic counter, zeroed by crf_init_kernel.
    // All threads fence (agent-scope cache inv) after the pass so every CU's
    // L0 sees the freshly written alpha buffer.
    auto gbarrier = [&](unsigned long long tgt) {
        __threadfence();
        __syncthreads();
        if (tid == 0) {
            __hip_atomic_fetch_add(bar, 1ull, __ATOMIC_ACQ_REL, __HIP_MEMORY_SCOPE_AGENT);
            while (__hip_atomic_load(bar, __ATOMIC_ACQUIRE, __HIP_MEMORY_SCOPE_AGENT) < tgt)
                __builtin_amdgcn_s_sleep(1);
        }
        __syncthreads();
        __threadfence();
    };

    gbarrier(FWD_WGS * nb); nb++;

    for (int t = 1; t < SEQ_LEN; ++t) {
        const float* aCur = alphaG + ((t - 1) & 1) * N_STATES;
        float*       aNxt = alphaG + (t & 1) * N_STATES;

        // ---- m = max(alpha) (shfl butterfly per wave, 1 block barrier) ----
        float lm = -__builtin_inff();
        for (int i = tid; i < N_STATES; i += FWD_THREADS) {
            float a = aCur[i];
            sA[i] = a;
            lm = fmaxf(lm, a);
        }
#pragma unroll
        for (int off = 16; off > 0; off >>= 1)
            lm = fmaxf(lm, __shfl_xor(lm, off, 32));
        if (lane == 0) sRed[wave] = lm;
        __syncthreads();
        const float m = fmaxf(fmaxf(sRed[0], sRed[1]), fmaxf(sRed[2], sRed[3]));

        // ---- v = exp(alpha - m) in fp16 ----
        for (int i = tid; i < N_STATES; i += FWD_THREADS)
            sV[i] = (_Float16)__expf(sA[i] - m);
        __syncthreads();

        // ---- u = v @ E-slab via WMMA; wave w handles K quarter w ----
        v8f acc;
#pragma unroll
        for (int q = 0; q < 8; ++q) acc[q] = 0.f;

        const int kk0   = wave * 8;
        const int hiOff = (lane >= 16) ? 16 : 0;
#pragma unroll
        for (int kk = kk0; kk < kk0 + 8; ++kk) {
            // A (16x32 fp16), v broadcast to all 16 rows:
            //  lane<16 : K = kb..kb+15  (kb = kk*32)
            //  lane>=16: shifted by +16
            int kb = kk * 32 + hiOff;
            v8h lo = *(const v8h*)&sV[kb];
            v8h hi = *(const v8h*)&sV[kb + 8];
            v16h a;
#pragma unroll
            for (int q = 0; q < 8; ++q) { a[q] = lo[q]; a[q + 8] = hi[q]; }
            v16h b = *(const v16h*)&sB[(kk * 32 + lane) * 16];
            acc = __builtin_amdgcn_wmma_f32_16x16x32_f16(
                false, a, false, b, (short)0, acc, false, false);
        }
        // every D row equals u (A rows identical); row M=0 lives in acc[0],
        // lanes 0..15 give N=0..15
        if (lane < 16) sPart[wave][lane] = acc[0];
        __syncthreads();

        // ---- alpha'[j] = log(u) + m + em[t][j] ----
        if (tid < COLS_PER_WG) {
            float u = sPart[0][tid] + sPart[1][tid] + sPart[2][tid] + sPart[3][tid];
            int j = cbase + tid;
            float a = (u > 0.f ? __logf(u) + m : -__builtin_inff())
                      + em[(size_t)t * N_STATES + j];
            aNxt[j] = a;
        }
        gbarrier(FWD_WGS * nb); nb++;
    }

    // ---- final: log_den = logsumexp(alpha_{T-1}); out = log_den - log_num ----
    if (wg == 0) {
        const float* aF = alphaG + ((SEQ_LEN - 1) & 1) * N_STATES;
        float lm = -__builtin_inff();
        for (int i = tid; i < N_STATES; i += FWD_THREADS) lm = fmaxf(lm, aF[i]);
#pragma unroll
        for (int off = 16; off > 0; off >>= 1)
            lm = fmaxf(lm, __shfl_xor(lm, off, 32));
        if (lane == 0) sRed[wave] = lm;
        __syncthreads();
        const float m = fmaxf(fmaxf(sRed[0], sRed[1]), fmaxf(sRed[2], sRed[3]));

        float ls = 0.f;
        for (int i = tid; i < N_STATES; i += FWD_THREADS) ls += __expf(aF[i] - m);
#pragma unroll
        for (int off = 16; off > 0; off >>= 1)
            ls += __shfl_xor(ls, off, 32);
        __syncthreads();
        if (lane == 0) sRed[wave] = ls;
        __syncthreads();
        if (tid == 0)
            out[0] = m + __logf(sRed[0] + sRed[1] + sRed[2] + sRed[3]) - ws_num[0];
    }
}

// ---------------------------------------------------------------------------
// Host launcher
// ---------------------------------------------------------------------------
extern "C" void kernel_launch(void* const* d_in, const int* in_sizes, int n_in,
                              void* d_out, int out_size, void* d_ws, size_t ws_size,
                              hipStream_t stream) {
    (void)in_sizes; (void)n_in; (void)out_size; (void)ws_size;

    const float* start      = (const float*)d_in[0];   // [1024]
    const float* transition = (const float*)d_in[1];   // [1024,1024]
    const float* emission   = (const float*)d_in[2];   // [1024,32000]
    const int*   obs_seq    = (const int*)d_in[3];     // [4096]
    const int*   state_seq  = (const int*)d_in[4];     // [4096]
    float*       out        = (float*)d_out;           // [1]

    // workspace layout
    char* ws = (char*)d_ws;
    unsigned long long* bar = (unsigned long long*)ws;               // 8 B
    float*     ws_num = (float*)(ws + 64);                           // 4 B
    float*     alphaG = (float*)(ws + 128);                          // 2*1024*4 = 8 KB
    _Float16*  E      = (_Float16*)(ws + 128 + 8192);                // 2 MB
    float*     em     = (float*)(ws + 128 + 8192 +
                                 (size_t)N_STATES * N_STATES * 2);   // 16 MB

    crf_init_kernel<<<1, 32, 0, stream>>>(bar);
    crf_exp_kernel<<<(N_STATES * N_STATES) / 256, 256, 0, stream>>>(transition, E);
    crf_gather_kernel<<<SEQ_LEN, 256, 0, stream>>>(emission, obs_seq, em);
    crf_numer_kernel<<<1, 256, 0, stream>>>(start, transition, emission,
                                            obs_seq, state_seq, ws_num);
    crf_forward_kernel<<<FWD_WGS, FWD_THREADS, 0, stream>>>(
        start, E, em, ws_num, bar, alphaG, out);
}
